// MultiHeadAttention_74071005987534
// MI455X (gfx1250) — matrix operand order repaired and checked by reading, not compiled, not run on hardware
//
#include <hip/hip_runtime.h>

// ---------------------------------------------------------------------------
// MI455X (gfx1250) multi-head attention, bf16 WMMA end-to-end.
//   B=2, N=2048, C=1024, H=16, D=64.
//   - softmax scale (1/sqrt(64)) and log2(e) folded into Q projection epilogue
//     -> flash loop uses raw v_exp_f32 (exp2), no S scaling
//   - softmax row-sum computed by WMMA (P x ones) -> no sum butterfly
// ---------------------------------------------------------------------------

typedef __bf16 v16bf __attribute__((ext_vector_type(16)));
typedef float  v8f   __attribute__((ext_vector_type(8)));
typedef unsigned int uif4 __attribute__((ext_vector_type(4)));

#define C_DIM  1024
#define HEADS  16
#define DHEAD  64
#define SEQ    2048
#define TOKENS 4096   // B * SEQ

__device__ __forceinline__ unsigned short f2bf(float f) {
  union { float f; unsigned u; } cv; cv.f = f;
  unsigned u = cv.u;
  unsigned r = u + 0x7FFFu + ((u >> 16) & 1u);   // round-to-nearest-even
  return (unsigned short)(r >> 16);
}

__device__ __forceinline__ float fast_exp2(float x) {
  return __builtin_amdgcn_exp2f(x);              // raw v_exp_f32
}

union FragU {
  v16bf v;
  uif4  q[2];
  unsigned short s[16];
};

// ---------------------------------------------------------------------------
// fp32 -> bf16 conversion (8 elems/thread, vectorized store)
// ---------------------------------------------------------------------------
__global__ __launch_bounds__(256) void cvt_f32_to_bf16(
    const float* __restrict__ in, unsigned short* __restrict__ out, int n8) {
  int i = blockIdx.x * 256 + threadIdx.x;
  if (i >= n8) return;
  const float* p = in + (size_t)i * 8;
  union { uif4 v; unsigned short s[8]; } o;
#pragma unroll
  for (int k = 0; k < 8; ++k) o.s[k] = f2bf(p[k]);
  *(uif4*)(out + (size_t)i * 8) = o.v;
}

// ---------------------------------------------------------------------------
// bf16 GEMM:  Y[M,1024] = (A[M,1024] @ W[1024,1024]^T + bias) * outScale
//   128x128 output tile per 256-thread (8-wave) workgroup.
//   Wave (wm,wn) owns a 64x32 slab = 4x2 grid of 16x16 WMMA tiles.
//   MODE 0: bf16 output scattered into [B, H, SEQ, DHEAD] head-major layout
//   MODE 1: fp32 output, row-major [M, 1024]
// ---------------------------------------------------------------------------
template <int MODE>
__global__ __launch_bounds__(256) void gemm_bf16_128x128(
    const unsigned short* __restrict__ A,   // [M,1024] bf16 row-major
    const unsigned short* __restrict__ W,   // [1024,1024] bf16 (out-channel major)
    const float* __restrict__ bias,         // [1024]
    void* __restrict__ outp, float outScale) {
  constexpr int K   = C_DIM;
  constexpr int LDT = 48;                    // padded LDS row stride (halves), 96B
  __shared__ alignas(16) unsigned short As[128 * LDT];
  __shared__ alignas(16) unsigned short Ws[128 * LDT];

  const int t     = threadIdx.x;
  const int lane  = t & 31;
  const int wave  = t >> 5;
  const int wm    = wave >> 2;               // 0..1  -> 64-row slab
  const int wn    = wave & 3;                // 0..3  -> 32-col slab
  const int blockM = blockIdx.y * 128;
  const int blockN = blockIdx.x * 128;

  const int lr = t >> 1;                     // tile-load row   0..127
  const int lc = (t & 1) * 16;               // tile-load col   0 or 16
  const int mlane = lane & 15;
  const int hlane = lane >> 4;

  v8f acc[4][2];
#pragma unroll
  for (int mi = 0; mi < 4; ++mi)
#pragma unroll
    for (int ni = 0; ni < 2; ++ni)
#pragma unroll
      for (int r = 0; r < 8; ++r) acc[mi][ni][r] = 0.0f;

  for (int kt = 0; kt < K / 32; ++kt) {
    // --- stage 128x32 A-tile and 128x32 W-tile (32B contiguous per thread) ---
    const uif4* ga = (const uif4*)(A + (size_t)(blockM + lr) * K + kt * 32 + lc);
    const uif4* gw = (const uif4*)(W + (size_t)(blockN + lr) * K + kt * 32 + lc);
    uif4 a0 = ga[0], a1 = ga[1];
    uif4 w0 = gw[0], w1 = gw[1];
    if (kt + 1 < K / 32) {                   // global_prefetch_b8 of next K-slice
      __builtin_prefetch(A + (size_t)(blockM + lr) * K + (kt + 1) * 32 + lc, 0, 0);
      __builtin_prefetch(W + (size_t)(blockN + lr) * K + (kt + 1) * 32 + lc, 0, 0);
    }
    *(uif4*)(As + lr * LDT + lc)     = a0;
    *(uif4*)(As + lr * LDT + lc + 8) = a1;
    *(uif4*)(Ws + lr * LDT + lc)     = w0;
    *(uif4*)(Ws + lr * LDT + lc + 8) = w1;
    __syncthreads();

    // --- A fragments: lane = row m, halves: K = b0..b0+7 and b0+16..b0+23 ---
    v16bf af[4], bfr[2];
#pragma unroll
    for (int mi = 0; mi < 4; ++mi) {
      const unsigned short* p = As + (wm * 64 + mi * 16 + mlane) * LDT + hlane * 8;
      FragU f;
      f.q[0] = *(const uif4*)(p);
      f.q[1] = *(const uif4*)(p + 16);
      af[mi] = f.v;
    }
    // --- B fragments: lane = column n (W row), K halves split by lane group ---
#pragma unroll
    for (int ni = 0; ni < 2; ++ni) {
      const unsigned short* p = Ws + (wn * 32 + ni * 16 + mlane) * LDT + hlane * 8;
      FragU f;
      f.q[0] = *(const uif4*)(p);
      f.q[1] = *(const uif4*)(p + 16);
      bfr[ni] = f.v;
    }
#pragma unroll
    for (int mi = 0; mi < 4; ++mi)
#pragma unroll
      for (int ni = 0; ni < 2; ++ni)
        acc[mi][ni] = __builtin_amdgcn_wmma_f32_16x16x32_bf16(
            false, af[mi], false, bfr[ni], (short)0, acc[mi][ni], false, false);
    __syncthreads();
  }

  // --- epilogue: C/D layout row = r + 8*(lane>>4), col = lane&15 ---
#pragma unroll
  for (int mi = 0; mi < 4; ++mi) {
#pragma unroll
    for (int ni = 0; ni < 2; ++ni) {
      const int col  = blockN + wn * 32 + ni * 16 + mlane;
      const float bc = bias[col];
#pragma unroll
      for (int r = 0; r < 8; ++r) {
        const int row = blockM + wm * 64 + mi * 16 + r + 8 * hlane;
        const float v = (acc[mi][ni][r] + bc) * outScale;
        if (MODE == 0) {
          unsigned short* o = (unsigned short*)outp;
          const int b = row >> 11, n = row & 2047;
          const int h = col >> 6,  d = col & 63;
          o[(((size_t)(b * HEADS + h)) * SEQ + n) * DHEAD + d] = f2bf(v);
        } else {
          float* o = (float*)outp;
          o[(size_t)row * C_DIM + col] = v;
        }
      }
    }
  }
}

// ---------------------------------------------------------------------------
// Flash attention: one (head, 64-query-row) block per 128-thread (4-wave) WG.
//   Wave w owns query rows [w*16, w*16+16). K staged row-major in LDS,
//   V staged TRANSPOSED (Vt[d][m]) so both K- and V-side B-fragments are two
//   contiguous ds_load_b128 per lane. Q carries 1/sqrt(d)*log2(e), so
//   softmax = exp2(s - m). Row-sum L is accumulated by WMMA (P x ones) with
//   the same alpha rescale as O -> no cross-lane sum reduction at all.
// ---------------------------------------------------------------------------
__global__ __launch_bounds__(128) void flash_attn_bf16(
    const unsigned short* __restrict__ Q,    // [B*H, SEQ, DHEAD] bf16 (pre-scaled)
    const unsigned short* __restrict__ Kg,
    const unsigned short* __restrict__ Vg,
    unsigned short* __restrict__ attb) {     // [TOKENS, C] bf16
  constexpr int LDK = 80;                    // K/P padded row stride (halves)
  constexpr int LDV = 72;                    // Vt padded row stride (144B = 9*16)
  __shared__ alignas(16) unsigned short Kt[64 * LDK];
  __shared__ alignas(16) unsigned short Vt[64 * LDV];   // transposed: [d][m]
  __shared__ alignas(16) unsigned short Pt[4 * 16 * LDK];

  const int bh   = blockIdx.x;               // b*H + h
  const int qb   = blockIdx.y;               // query block (64 rows)
  const int t    = threadIdx.x;
  const int lane = t & 31;
  const int wave = t >> 5;
  const int mlane = lane & 15;
  const int hlane = lane >> 4;

  // Q fragments held in registers for the whole key loop (2 K-chunks of 32)
  v16bf aQ[2];
  {
    const unsigned short* qrow =
        Q + ((size_t)bh * SEQ + qb * 64 + wave * 16 + mlane) * DHEAD;
#pragma unroll
    for (int c = 0; c < 2; ++c) {
      FragU f;
      f.q[0] = *(const uif4*)(qrow + c * 32 + hlane * 8);
      f.q[1] = *(const uif4*)(qrow + c * 32 + hlane * 8 + 16);
      aQ[c] = f.v;
    }
  }

  // all-ones bf16 B-fragment for the row-sum WMMA
  FragU fones;
#pragma unroll
  for (int i = 0; i < 16; ++i) fones.s[i] = 0x3F80;    // bf16 1.0

  v8f o[4], ls;                               // O accum + row-sum accum
#pragma unroll
  for (int j = 0; j < 4; ++j)
#pragma unroll
    for (int r = 0; r < 8; ++r) o[j][r] = 0.0f;
#pragma unroll
  for (int r = 0; r < 8; ++r) ls[r] = 0.0f;
  float mrun[8];
#pragma unroll
  for (int r = 0; r < 8; ++r) mrun[r] = -3.0e38f;

  for (int kb = 0; kb < SEQ / 64; ++kb) {
    // --- stage K row-major, V transposed (coalesced 16B global loads) ---
#pragma unroll
    for (int it = 0; it < 4; ++it) {
      const int qi  = t + it * 128;          // 0..511
      const int row = qi >> 3, c16 = (qi & 7) * 8;
      const size_t g = ((size_t)bh * SEQ + kb * 64 + row) * DHEAD + c16;
      *(uif4*)(Kt + row * LDK + c16) = *(const uif4*)(Kg + g);
      union { uif4 v; unsigned short s[8]; } vv;
      vv.v = *(const uif4*)(Vg + g);
#pragma unroll
      for (int e = 0; e < 8; ++e)            // scatter: Vt[d][m] = V[m][d]
        Vt[(c16 + e) * LDV + row] = vv.s[e];
    }
    __syncthreads();

    // --- S = Q K^T (log2-domain) : B-fragment column n = key row (j*16+n) ---
    v8f s[4];
#pragma unroll
    for (int j = 0; j < 4; ++j)
#pragma unroll
      for (int r = 0; r < 8; ++r) s[j][r] = 0.0f;
#pragma unroll
    for (int c = 0; c < 2; ++c) {
#pragma unroll
      for (int j = 0; j < 4; ++j) {
        const unsigned short* p = Kt + (j * 16 + mlane) * LDK + c * 32 + hlane * 8;
        FragU f;
        f.q[0] = *(const uif4*)(p);
        f.q[1] = *(const uif4*)(p + 16);
        s[j] = __builtin_amdgcn_wmma_f32_16x16x32_bf16(
            false, aQ[c], false, f.v, (short)0, s[j], false, false);
      }
    }

    // --- online softmax (base-2); row r lives in one (lane>>4) group ---
    unsigned short* pw = Pt + wave * 16 * LDK;    // wave-private P tile
#pragma unroll
    for (int r = 0; r < 8; ++r) {
      float mloc = fmaxf(fmaxf(s[0][r], s[1][r]), fmaxf(s[2][r], s[3][r]));
#pragma unroll
      for (int msk = 1; msk < 16; msk <<= 1)
        mloc = fmaxf(mloc, __shfl_xor(mloc, msk, 32));
      const float mnew  = fmaxf(mrun[r], mloc);
      const float alpha = fast_exp2(mrun[r] - mnew);
      mrun[r] = mnew;
#pragma unroll
      for (int j = 0; j < 4; ++j)
        pw[(r + 8 * hlane) * LDK + j * 16 + mlane] = f2bf(fast_exp2(s[j][r] - mnew));
#pragma unroll
      for (int j = 0; j < 4; ++j) o[j][r] *= alpha;
      ls[r] *= alpha;
    }

    // --- O += P V ; L += P 1  (DS in-order within a wave -> no barrier) ---
#pragma unroll
    for (int c2 = 0; c2 < 2; ++c2) {
      FragU fp;                               // P A-fragment (wave-private tile)
      const unsigned short* pp = pw + mlane * LDK + c2 * 32 + hlane * 8;
      fp.q[0] = *(const uif4*)(pp);
      fp.q[1] = *(const uif4*)(pp + 16);
#pragma unroll
      for (int j = 0; j < 4; ++j) {
        // V B-fragment from transposed tile: column n = d, contiguous in m
        const unsigned short* pv = Vt + (j * 16 + mlane) * LDV + c2 * 32 + hlane * 8;
        FragU fv;
        fv.q[0] = *(const uif4*)(pv);
        fv.q[1] = *(const uif4*)(pv + 16);
        o[j] = __builtin_amdgcn_wmma_f32_16x16x32_bf16(
            false, fp.v, false, fv.v, (short)0, o[j], false, false);
      }
      ls = __builtin_amdgcn_wmma_f32_16x16x32_bf16(
          false, fp.v, false, fones.v, (short)0, ls, false, false);
    }
    __syncthreads();                          // Kt/Vt reuse next iteration
  }

  // --- normalize (L replicated in every lane column) and scatter ---
  const int b = bh >> 4, h = bh & 15;
#pragma unroll
  for (int r = 0; r < 8; ++r) {
    const float inv = 1.0f / ls[r];
    const int row = qb * 64 + wave * 16 + r + 8 * hlane;
#pragma unroll
    for (int j = 0; j < 4; ++j)
      attb[((size_t)(b * SEQ + row)) * C_DIM + h * DHEAD + j * 16 + mlane] =
          f2bf(o[j][r] * inv);
  }
}

// ---------------------------------------------------------------------------
// Launch: cvt x + 4 weights -> bf16; 3 projection GEMMs -> head-major Q/K/V
//         (Q pre-scaled by 1/sqrt(d)*log2(e)); flash attention; output
//         projection GEMM -> fp32 d_out.  48 MB bf16 workspace (fits in L2).
// ---------------------------------------------------------------------------
extern "C" void kernel_launch(void* const* d_in, const int* in_sizes, int n_in,
                              void* d_out, int out_size, void* d_ws, size_t ws_size,
                              hipStream_t stream) {
  (void)in_sizes; (void)n_in; (void)out_size; (void)ws_size;
  const float* x  = (const float*)d_in[0];
  const float* Wq = (const float*)d_in[1];
  const float* bq = (const float*)d_in[2];
  const float* Wk = (const float*)d_in[3];
  const float* bk = (const float*)d_in[4];
  const float* Wv = (const float*)d_in[5];
  const float* bv = (const float*)d_in[6];
  const float* Wo = (const float*)d_in[7];
  const float* bo = (const float*)d_in[8];

  unsigned short* ws  = (unsigned short*)d_ws;
  unsigned short* xb  = ws;                                    // [4096,1024]
  unsigned short* wqb = xb  + (size_t)TOKENS * C_DIM;          // [1024,1024] each
  unsigned short* wkb = wqb + (size_t)C_DIM * C_DIM;
  unsigned short* wvb = wkb + (size_t)C_DIM * C_DIM;
  unsigned short* wob = wvb + (size_t)C_DIM * C_DIM;
  unsigned short* qh  = wob + (size_t)C_DIM * C_DIM;           // [32,2048,64] each
  unsigned short* kh  = qh  + (size_t)32 * SEQ * DHEAD;
  unsigned short* vh  = kh  + (size_t)32 * SEQ * DHEAD;
  unsigned short* ab  = vh  + (size_t)32 * SEQ * DHEAD;        // [4096,1024]

  const int nX = TOKENS * C_DIM / 8, nW = C_DIM * C_DIM / 8;
  cvt_f32_to_bf16<<<(nX + 255) / 256, 256, 0, stream>>>(x,  xb,  nX);
  cvt_f32_to_bf16<<<(nW + 255) / 256, 256, 0, stream>>>(Wq, wqb, nW);
  cvt_f32_to_bf16<<<(nW + 255) / 256, 256, 0, stream>>>(Wk, wkb, nW);
  cvt_f32_to_bf16<<<(nW + 255) / 256, 256, 0, stream>>>(Wv, wvb, nW);
  cvt_f32_to_bf16<<<(nW + 255) / 256, 256, 0, stream>>>(Wo, wob, nW);

  const float qScale = 0.125f * 1.4426950408889634f;           // 1/sqrt(64)*log2(e)
  dim3 gg(C_DIM / 128, TOKENS / 128);                          // (8, 32)
  gemm_bf16_128x128<0><<<gg, 256, 0, stream>>>(xb, wqb, bq, qh, qScale);
  gemm_bf16_128x128<0><<<gg, 256, 0, stream>>>(xb, wkb, bk, kh, 1.0f);
  gemm_bf16_128x128<0><<<gg, 256, 0, stream>>>(xb, wvb, bv, vh, 1.0f);

  dim3 ga(32, SEQ / 64);                                       // (32, 32)
  flash_attn_bf16<<<ga, 128, 0, stream>>>(qh, kh, vh, ab);

  gemm_bf16_128x128<1><<<gg, 256, 0, stream>>>(ab, wob, bo, d_out, 1.0f);
}
